// DocRedModel_40183714022097
// MI455X (gfx1250) — matrix operand order repaired and checked by reading, not compiled, not run on hardware
//
#include <hip/hip_runtime.h>

// ---------------- problem constants ----------------
constexpr int B_ = 4, M_ = 128, H_ = 1024, NH_ = 16, L_ = 1024;
constexpr int E_ = 64, R_ = 512, EMB_ = 768, BS_ = 64, NC_ = 97;
constexpr int N_   = B_ * R_;      // 2048 pairs total
constexpr int K1_  = 2 * H_;       // 2048 (concat dim)
constexpr int KB_  = EMB_ * BS_;   // 49152 (bilinear K)
constexpr int NCP_ = 128;          // NC padded to 4*32 (32-wide n tiles)

typedef __attribute__((ext_vector_type(16))) __bf16 v16bf;
typedef __attribute__((ext_vector_type(8)))  __bf16 v8bf;
typedef __attribute__((ext_vector_type(8)))  float  v8f;
typedef __attribute__((ext_vector_type(4)))  unsigned int u32x4;
typedef __attribute__((ext_vector_type(8)))  int i32x8;
typedef __attribute__((ext_vector_type(4)))  int i32x4;

// GEMM tiling: 256 threads = 8 wave32; block tile 128(m) x 32(n), K-step 32.
// Each wave: one 16-row A fragment, two B fragments, two WMMAs per k-step.
// LDS tiles are double-buffered; TDM loads for step s+1 are issued before the
// wait for step s (TENSORcnt decrements in order, so tensorcnt<=2 / <=1 proves
// the current buffer has landed while the next buffer's DMA is still in flight).
constexpr int TM_  = 128;
constexpr int TN_  = 32;
constexpr int TK_  = 32;
constexpr int ASTR = 40;   // 80B row stride: multiple of 16B, 20-bank stride
constexpr int BSTR = 40;

#if defined(__HIP_DEVICE_COMPILE__) && __has_builtin(__builtin_amdgcn_tensor_load_to_lds)
#define HAS_TDM 1
#else
#define HAS_TDM 0
#endif

__device__ inline unsigned lds_off(const void* p)
{
    // Generic pointers into LDS carry the LDS byte offset in bits [31:0] (aperture rule).
    return (unsigned)(unsigned long long)p;
}

__device__ inline void wait_tensorcnt0()
{
#if __has_builtin(__builtin_amdgcn_s_wait_tensorcnt)
    __builtin_amdgcn_s_wait_tensorcnt(0);
#else
    asm volatile("s_wait_tensorcnt 0x0" ::: "memory");
#endif
}
__device__ inline void wait_tensorcnt1()
{
#if __has_builtin(__builtin_amdgcn_s_wait_tensorcnt)
    __builtin_amdgcn_s_wait_tensorcnt(1);
#else
    asm volatile("s_wait_tensorcnt 0x1" ::: "memory");
#endif
}
__device__ inline void wait_tensorcnt2()
{
#if __has_builtin(__builtin_amdgcn_s_wait_tensorcnt)
    __builtin_amdgcn_s_wait_tensorcnt(2);
#else
    asm volatile("s_wait_tensorcnt 0x2" ::: "memory");
#endif
}

#if HAS_TDM
// TDM: load a [rows x 32] bf16 tile (row stride K elements) from global into LDS,
// with hardware padding 16 DWORDs data + 4 DWORDs pad => 80B LDS row stride (ASTR=40).
__device__ inline void tdm_load_tile(const __bf16* g, unsigned lds_addr, int K, int rows)
{
    const unsigned long long ga = (unsigned long long)g;
    u32x4 g0;
    g0.x = 1u;                                                  // count=1, user descriptor
    g0.y = lds_addr;                                            // LDS byte address
    g0.z = (unsigned)ga;                                        // global_addr[31:0]
    g0.w = ((unsigned)(ga >> 32) & 0x01FFFFFFu) | (2u << 30);   // addr[56:32], type=2 (image)
    i32x8 g1;
    g1[0] = (1 << 16)                                           // data_size = 2 bytes
          | (1 << 20)                                           // pad_enable
          | (3 << 22)                                           // pad_interval: 16 DWORDs
          | (3 << 25);                                          // pad_amount: 4 DWORDs
    g1[1] = (int)(((unsigned)K & 0xFFFFu) << 16);               // tensor_dim0 low16 @ [63:48]
    g1[2] = (int)(((unsigned)K >> 16) |                         // tensor_dim0 high16
                  (((unsigned)rows & 0xFFFFu) << 16));          // tensor_dim1 low16
    g1[3] = (int)(32u << 16);                                   // tensor_dim1 hi=0 | tile_dim0=32
    g1[4] = rows & 0xFFFF;                                      // tile_dim1 | tile_dim2=0
    g1[5] = K;                                                  // tensor_dim0_stride low32
    g1[6] = 0;                                                  // stride hi | dim1_stride low
    g1[7] = 0;
    const i32x4 z4 = {0, 0, 0, 0};
#if __clang_major__ >= 23
    const i32x8 z8 = {0, 0, 0, 0, 0, 0, 0, 0};
    __builtin_amdgcn_tensor_load_to_lds(g0, g1, z4, z4, z8, 0);
#else
    __builtin_amdgcn_tensor_load_to_lds(g0, g1, z4, z4, 0);
#endif
}
#endif

// ---------------- mention -> entity pooling (segment logsumexp) ----------------
__global__ __launch_bounds__(256)
void pool_entities_kernel(const float* __restrict__ ent_lhs,   // [B, M, H]
                          const int*   __restrict__ labels,    // [B, M]
                          float* __restrict__ ent_embeds,      // [B*E, H]
                          float* __restrict__ cnt)             // [B*E]
{
    const int seg = blockIdx.x;            // b*E + e
    const int b = seg / E_, e = seg % E_;
    const int t = threadIdx.x;
    __shared__ int lab[M_];
    if (t < M_) lab[t] = labels[b * M_ + t];
    __syncthreads();

    float mx[4];
#pragma unroll
    for (int q = 0; q < 4; ++q) mx[q] = -3.0e38f;
    int c = 0;
    for (int m = 0; m < M_; ++m) {
        if (lab[m] == e) {
            ++c;
            const float* row = ent_lhs + ((size_t)(b * M_ + m)) * H_;
#pragma unroll
            for (int q = 0; q < 4; ++q) mx[q] = fmaxf(mx[q], row[q * 256 + t]);
        }
    }
    float se[4] = {0.f, 0.f, 0.f, 0.f};
    for (int m = 0; m < M_; ++m) {
        if (lab[m] == e) {
            const float* row = ent_lhs + ((size_t)(b * M_ + m)) * H_;
#pragma unroll
            for (int q = 0; q < 4; ++q) se[q] += expf(row[q * 256 + t] - mx[q]);
        }
    }
#pragma unroll
    for (int q = 0; q < 4; ++q) {
        float v = 0.f;
        if (c > 0) v = mx[q] + logf(fmaxf(se[q], 1e-30f));
        ent_embeds[(size_t)seg * H_ + q * 256 + t] = v;
    }
    if (t == 0) cnt[seg] = (float)c;
}

// ---------------- mention -> entity attention mean ----------------
__global__ __launch_bounds__(256)
void pool_attn_kernel(const float* __restrict__ attn,     // [B, NH, M, L]
                      const int*   __restrict__ labels,   // [B, M]
                      const float* __restrict__ cnt,      // [B*E]
                      float* __restrict__ ent_attn)       // [B*E, NH, L]
{
    const int seg = blockIdx.x;            // b*E + e
    const int b = seg / E_, e = seg % E_;
    const int nh = blockIdx.y;
    const int t = threadIdx.x;
    __shared__ int lab[M_];
    if (t < M_) lab[t] = labels[b * M_ + t];
    __syncthreads();

    float s[4] = {0.f, 0.f, 0.f, 0.f};
    for (int m = 0; m < M_; ++m) {
        if (lab[m] == e) {
            const float* row = attn + ((size_t)((b * NH_ + nh) * M_ + m)) * L_;
#pragma unroll
            for (int q = 0; q < 4; ++q) s[q] += row[q * 256 + t];
        }
    }
    const float inv = 1.f / fmaxf(cnt[seg], 1.f);
    float* out = ent_attn + ((size_t)seg * NH_ + nh) * L_;
#pragma unroll
    for (int q = 0; q < 4; ++q) out[q * 256 + t] = s[q] * inv;
}

// ---------------- per-pair attention product + normalize -> bf16 ----------------
__global__ __launch_bounds__(256)
void ht_attn_kernel(const float* __restrict__ ent_attn,   // [B*E, NH, L]
                    const int*   __restrict__ hts,        // [B, R, 2]
                    __bf16* __restrict__ ht_out)          // [B*R, L]
{
    const int p = blockIdx.x;              // b*R + r
    const int b = p / R_;
    const int hi = hts[p * 2 + 0];
    const int ti = hts[p * 2 + 1];
    const float* ha = ent_attn + ((size_t)(b * E_ + hi)) * NH_ * L_;
    const float* ta = ent_attn + ((size_t)(b * E_ + ti)) * NH_ * L_;
    const int t = threadIdx.x;

    float v[4];
    float psum = 0.f;
#pragma unroll
    for (int q = 0; q < 4; ++q) {
        const int l = q * 256 + t;
        float s = 0.f;
#pragma unroll
        for (int nh = 0; nh < NH_; ++nh) s += ha[nh * L_ + l] * ta[nh * L_ + l];
        v[q] = s * (1.f / NH_);
        psum += v[q];
    }
    __shared__ float red[256];
    red[t] = psum;
    __syncthreads();
    for (int o = 128; o > 0; o >>= 1) {
        if (t < o) red[t] += red[t + o];
        __syncthreads();
    }
    const float inv = 1.f / (red[0] + 1e-5f);
#pragma unroll
    for (int q = 0; q < 4; ++q)
        ht_out[(size_t)p * L_ + q * 256 + t] = (__bf16)(v[q] * inv);
}

// ---------------- f32 [Krows x Ncols] -> bf16 transposed [Npad x Krows] ----------------
__global__ __launch_bounds__(256)
void transpose_bf16_kernel(const float* __restrict__ in, __bf16* __restrict__ out,
                           int Krows, int Ncols, int Npad,
                           long long in_bstride, long long out_bstride)
{
    long long idx = (long long)blockIdx.x * 256 + threadIdx.x;  // over Npad*Krows
    const long long total = (long long)Npad * Krows;
    if (idx >= total) return;
    const int c = (int)(idx / Krows);      // output row (col of input)
    const int r = (int)(idx % Krows);
    const float* ip = in + (long long)blockIdx.y * in_bstride;
    __bf16* op = out + (long long)blockIdx.y * out_bstride;
    float v = (c < Ncols) ? ip[(size_t)r * Ncols + c] : 0.f;
    op[idx] = (__bf16)v;
}

// ---------------- build concat A matrices: Ah/At = [gather(ent), rel] ----------------
__global__ __launch_bounds__(256)
void build_A_kernel(const float* __restrict__ ent_embeds,  // [B*E, H]
                    const __bf16* __restrict__ rel,        // [N, H]
                    const int*   __restrict__ hts,         // [B, R, 2]
                    __bf16* __restrict__ Ah, __bf16* __restrict__ At)
{
    long long idx = (long long)blockIdx.x * 256 + threadIdx.x;  // over N*2048
    if (idx >= (long long)N_ * K1_) return;
    const int n = (int)(idx >> 11);
    const int k = (int)(idx & (K1_ - 1));
    if (k < H_) {
        const int b = n / R_;
        const int hi = hts[n * 2 + 0];
        const int ti = hts[n * 2 + 1];
        Ah[idx] = (__bf16)ent_embeds[((size_t)(b * E_ + hi)) * H_ + k];
        At[idx] = (__bf16)ent_embeds[((size_t)(b * E_ + ti)) * H_ + k];
    } else {
        const __bf16 v = rel[(size_t)n * H_ + (k - H_)];
        Ah[idx] = v;
        At[idx] = v;
    }
}

// ---------------- fragment helpers (ISA 16-bit WMMA layouts) ----------------
__device__ inline v16bf load_frag_a(const __bf16* As, int mrow, int lane)
{
    const int hi = (lane >> 4) & 1;               // lanes 16-31 take k+8 halves
    const __bf16* row = As + mrow * ASTR;
    const v8bf lo = *(const v8bf*)(row + hi * 8);        // k = 0..7 / 8..15
    const v8bf hh = *(const v8bf*)(row + 16 + hi * 8);   // k = 16..23 / 24..31
    v16bf a;
#pragma unroll
    for (int i = 0; i < 8; ++i) { a[i] = lo[i]; a[8 + i] = hh[i]; }
    return a;
}

__device__ inline v16bf load_frag_b(const __bf16* Bs, int lane)
{
    const int hi = (lane >> 4) & 1;               // lanes 16-31 take k+8 halves
    const __bf16* row = Bs + (lane & 15) * BSTR + hi * 8;
    const v8bf lo = *(const v8bf*)(row);
    const v8bf hh = *(const v8bf*)(row + 16);
    v16bf b;
#pragma unroll
    for (int i = 0; i < 8; ++i) { b[i] = lo[i]; b[8 + i] = hh[i]; }
    return b;
}

// ---------------- generic bf16 WMMA GEMM: C = act(A @ Bt^T + bias) ----------------
// A:  [Mtot, K] bf16 row-major.   Bt: [>=gridDim.y*32, K] bf16 (B transposed).
// flags: bit0 = f32 output (else bf16), bit1 = tanh activation.
__global__ __launch_bounds__(256)
void gemm_bf16_wmma(const __bf16* __restrict__ A, const __bf16* __restrict__ Bt,
                    void* __restrict__ Cv, const float* __restrict__ bias,
                    int Mtot, int Nstore, int K, int ldc, int flags,
                    long long sA, long long sB, long long sC)
{
    A  += (long long)blockIdx.z * sA;
    Bt += (long long)blockIdx.z * sB;

    __shared__ __align__(16) __bf16 As[2][TM_ * ASTR];
    __shared__ __align__(16) __bf16 Bs[2][TN_ * BSTR];

    const int t = threadIdx.x;
    const int lane = t & 31, wave = t >> 5;
    const int m_base = blockIdx.x * TM_;
    const int n_base = blockIdx.y * TN_;
    const int nsteps = K / TK_;

    v8f acc0 = {}, acc1 = {};

#if HAS_TDM
    if (wave == 0) {   // prologue: DMA first tiles into buffer 0
        tdm_load_tile(A + (size_t)m_base * K, lds_off(As[0]), K, TM_);
        tdm_load_tile(Bt + (size_t)n_base * K, lds_off(Bs[0]), K, TN_);
    }
#endif

    for (int s = 0; s < nsteps; ++s) {
        const int cur = s & 1;
        const int k0 = s * TK_;
#if HAS_TDM
        if (wave == 0) {
            if (s + 1 < nsteps) {   // issue next buffer's DMA, then prove cur landed
                const int nxt = cur ^ 1;
                tdm_load_tile(A + (size_t)m_base * K + k0 + TK_, lds_off(As[nxt]), K, TM_);
                tdm_load_tile(Bt + (size_t)n_base * K + k0 + TK_, lds_off(Bs[nxt]), K, TN_);
                wait_tensorcnt2();   // in-order: only next buffer's 2 loads outstanding
            } else {
                wait_tensorcnt0();
            }
        }
        __syncthreads();            // publish buffer cur to all waves
#else
        __syncthreads();
        // A tile: 128 rows x 32 k = 512 x 16B chunks
#pragma unroll
        for (int j = 0; j < 2; ++j) {
            const int id = t + 256 * j;
            const int row = id >> 2, ch = id & 3;
            const u32x4 v = *(const u32x4*)(A + (size_t)(m_base + row) * K + k0 + ch * 8);
            *(u32x4*)(As[cur] + row * ASTR + ch * 8) = v;
        }
        // B tile: 32 rows x 32 k = 128 x 16B chunks
        if (t < 128) {
            const int row = t >> 2, ch = t & 3;
            const u32x4 v = *(const u32x4*)(Bt + (size_t)(n_base + row) * K + k0 + ch * 8);
            *(u32x4*)(Bs[cur] + row * BSTR + ch * 8) = v;
        }
        __syncthreads();
#endif

        const v16bf af = load_frag_a(As[cur], wave * 16 + (lane & 15), lane);
        const v16bf b0 = load_frag_b(Bs[cur], lane);
        const v16bf b1 = load_frag_b(Bs[cur] + 16 * BSTR, lane);
        acc0 = __builtin_amdgcn_wmma_f32_16x16x32_bf16(false, af, false, b0,
                                                       (short)0, acc0, false, false);
        acc1 = __builtin_amdgcn_wmma_f32_16x16x32_bf16(false, af, false, b1,
                                                       (short)0, acc1, false, false);
#if HAS_TDM
        __syncthreads();   // all waves done reading cur before it is DMA-overwritten
#endif
    }

    // D layout: lane<16 -> (m=r, n=lane); lane>=16 -> (m=8+r, n=lane-16)
    const int nl = lane & 15;
    const int m_sub = (lane >> 4) * 8;
#pragma unroll
    for (int h = 0; h < 2; ++h) {
        const int n_g = n_base + h * 16 + nl;
        if (n_g >= Nstore) continue;
        const float bv = bias ? bias[n_g] : 0.f;
        const v8f accv = h ? acc1 : acc0;
#pragma unroll
        for (int r = 0; r < 8; ++r) {
            const int m_g = m_base + wave * 16 + m_sub + r;
            if (m_g >= Mtot) break;
            float v = accv[r] + bv;
            if (flags & 2) v = tanhf(v);
            if (flags & 1)
                ((float*)Cv + (long long)blockIdx.z * sC)[(long long)m_g * ldc + n_g] = v;
            else
                ((__bf16*)Cv + (long long)blockIdx.z * sC)[(long long)m_g * ldc + n_g] = (__bf16)v;
        }
    }
}

// ---------------- block-bilinear classifier GEMM, A formed on the fly ----------------
// logits[n,c] = sum_f hs[n,k64*64+i]*ts[n,k64*64+j] * Wb[f,c] + bb[c],
//   f = k64*4096 + i*64 + j.  Each 32-wide K tile has fixed (k64,i), j contiguous:
//   A[row, jj] = hs[n, k64*64+i] * ts[n, k64*64+j0+jj]  -> generated in LDS, never
//   materialized in global memory (full A would be 192MB).
__device__ inline void bilinear_gen_A(__bf16* dstA, const __bf16* Hs, const __bf16* Ts,
                                      int m_base, int k0, int t)
{
    const int k64 = k0 >> 12;          // / 4096
    const int rem = k0 & 4095;
    const int i  = rem >> 6;           // fixed i within this K tile
    const int j0 = rem & 63;           // 0 or 32
#pragma unroll
    for (int jj = 0; jj < 2; ++jj) {
        const int id = t + 256 * jj;
        const int row = id >> 2, ch = id & 3;
        const int n = m_base + row;
        const float hv = (float)Hs[(size_t)n * EMB_ + k64 * 64 + i];
        const v8bf tv = *(const v8bf*)(Ts + (size_t)n * EMB_ + k64 * 64 + j0 + ch * 8);
        v8bf av;
#pragma unroll
        for (int q = 0; q < 8; ++q) av[q] = (__bf16)(hv * (float)tv[q]);
        *(v8bf*)(dstA + row * ASTR + ch * 8) = av;
    }
}

__global__ __launch_bounds__(256)
void gemm_bilinear_wmma(const __bf16* __restrict__ Hs,   // [N, EMB]
                        const __bf16* __restrict__ Ts,   // [N, EMB]
                        const __bf16* __restrict__ WbT,  // [NCP, KB] (zero-padded rows)
                        const float*  __restrict__ bb,   // [NC]
                        float* __restrict__ out)         // [N, NC]
{
    __shared__ __align__(16) __bf16 As[2][TM_ * ASTR];
    __shared__ __align__(16) __bf16 Bs[2][TN_ * BSTR];

    const int t = threadIdx.x;
    const int lane = t & 31, wave = t >> 5;
    const int m_base = blockIdx.x * TM_;
    const int n_base = blockIdx.y * TN_;
    const int nsteps = KB_ / TK_;

    v8f acc0 = {}, acc1 = {};

    // prologue: B DMA + A generation for buffer 0
#if HAS_TDM
    if (wave == 0) tdm_load_tile(WbT + (size_t)n_base * KB_, lds_off(Bs[0]), KB_, TN_);
#else
    if (t < 128) {
        const int row = t >> 2, ch = t & 3;
        const u32x4 v = *(const u32x4*)(WbT + (size_t)(n_base + row) * KB_ + ch * 8);
        *(u32x4*)(Bs[0] + row * BSTR + ch * 8) = v;
    }
#endif
    bilinear_gen_A(As[0], Hs, Ts, m_base, 0, t);

    for (int s = 0; s < nsteps; ++s) {
        const int cur = s & 1;
        const int k0 = s * TK_;
        if (s + 1 < nsteps) {      // stage next buffer (DMA B, generate A)
            const int nxt = cur ^ 1;
#if HAS_TDM
            if (wave == 0)
                tdm_load_tile(WbT + (size_t)n_base * KB_ + k0 + TK_, lds_off(Bs[nxt]), KB_, TN_);
#else
            if (t < 128) {
                const int row = t >> 2, ch = t & 3;
                const u32x4 v = *(const u32x4*)(WbT + (size_t)(n_base + row) * KB_ + k0 + TK_ + ch * 8);
                *(u32x4*)(Bs[nxt] + row * BSTR + ch * 8) = v;
            }
#endif
            bilinear_gen_A(As[nxt], Hs, Ts, m_base, k0 + TK_, t);
#if HAS_TDM
            if (wave == 0) wait_tensorcnt1();   // in-order: cur's B load has landed
#endif
        } else {
#if HAS_TDM
            if (wave == 0) wait_tensorcnt0();
#endif
        }
        __syncthreads();           // publish buffer cur

        const v16bf af = load_frag_a(As[cur], wave * 16 + (lane & 15), lane);
        const v16bf b0 = load_frag_b(Bs[cur], lane);
        const v16bf b1 = load_frag_b(Bs[cur] + 16 * BSTR, lane);
        acc0 = __builtin_amdgcn_wmma_f32_16x16x32_bf16(false, af, false, b0,
                                                       (short)0, acc0, false, false);
        acc1 = __builtin_amdgcn_wmma_f32_16x16x32_bf16(false, af, false, b1,
                                                       (short)0, acc1, false, false);
        __syncthreads();           // readers done before cur is restaged
    }

    const int nl = lane & 15;
    const int m_sub = (lane >> 4) * 8;
#pragma unroll
    for (int h = 0; h < 2; ++h) {
        const int n_g = n_base + h * 16 + nl;
        if (n_g >= NC_) continue;
        const float bv = bb[n_g];
        const v8f accv = h ? acc1 : acc0;
#pragma unroll
        for (int r = 0; r < 8; ++r) {
            const int m_g = m_base + wave * 16 + m_sub + r;
            out[(size_t)m_g * NC_ + n_g] = accv[r] + bv;
        }
    }
}

// ---------------- host orchestration ----------------
extern "C" void kernel_launch(void* const* d_in, const int* in_sizes, int n_in,
                              void* d_out, int out_size, void* d_ws, size_t ws_size,
                              hipStream_t stream)
{
    const float* seq_lhs = (const float*)d_in[0];   // [B, L, H]
    const float* ent_lhs = (const float*)d_in[1];   // [B, M, H]
    const float* attn    = (const float*)d_in[2];   // [B, NH, M, L]
    const int*   labels  = (const int*)d_in[3];     // [B, M]
    const int*   hts     = (const int*)d_in[4];     // [B, R, 2]
    const float* Wh      = (const float*)d_in[5];   // [2H, EMB]
    const float* bh      = (const float*)d_in[6];   // [EMB]
    const float* Wt      = (const float*)d_in[7];   // [2H, EMB]
    const float* bt      = (const float*)d_in[8];   // [EMB]
    const float* Wb      = (const float*)d_in[9];   // [EMB*BS, NC]
    const float* bb      = (const float*)d_in[10];  // [NC]
    float* out = (float*)d_out;                     // [N, NC]

    char* ws = (char*)d_ws;
    size_t off = 0;
    auto take = [&](size_t bytes) { size_t o = off; off += (bytes + 255) & ~(size_t)255; return o; };

    float*  ent_embeds = (float*)(ws + take((size_t)B_ * E_ * H_ * 4));
    float*  cnt        = (float*)(ws + take((size_t)B_ * E_ * 4));
    float*  ent_attn   = (float*)(ws + take((size_t)B_ * E_ * NH_ * L_ * 4));
    __bf16* ht         = (__bf16*)(ws + take((size_t)N_ * L_ * 2));
    __bf16* seqT       = (__bf16*)(ws + take((size_t)B_ * H_ * L_ * 2));
    __bf16* WhT        = (__bf16*)(ws + take((size_t)EMB_ * K1_ * 2));
    __bf16* WtT        = (__bf16*)(ws + take((size_t)EMB_ * K1_ * 2));
    __bf16* WbT        = (__bf16*)(ws + take((size_t)NCP_ * KB_ * 2));
    __bf16* rel        = (__bf16*)(ws + take((size_t)N_ * H_ * 2));
    __bf16* Ah         = (__bf16*)(ws + take((size_t)N_ * K1_ * 2));
    __bf16* At         = (__bf16*)(ws + take((size_t)N_ * K1_ * 2));
    __bf16* hsO        = (__bf16*)(ws + take((size_t)N_ * EMB_ * 2));
    __bf16* tsO        = (__bf16*)(ws + take((size_t)N_ * EMB_ * 2));

    // 1. entity pooling (logsumexp) + counts
    pool_entities_kernel<<<B_ * E_, 256, 0, stream>>>(ent_lhs, labels, ent_embeds, cnt);
    // 2. entity attention means
    pool_attn_kernel<<<dim3(B_ * E_, NH_), 256, 0, stream>>>(attn, labels, cnt, ent_attn);
    // 3. pair attention product, normalize -> bf16
    ht_attn_kernel<<<N_, 256, 0, stream>>>(ent_attn, hts, ht);
    // 4. bf16 transposed copies of all GEMM B-operands
    transpose_bf16_kernel<<<dim3((H_ * L_) / 256, B_), 256, 0, stream>>>(
        seq_lhs, seqT, L_, H_, H_, (long long)L_ * H_, (long long)H_ * L_);
    transpose_bf16_kernel<<<dim3((EMB_ * K1_) / 256, 1), 256, 0, stream>>>(
        Wh, WhT, K1_, EMB_, EMB_, 0, 0);
    transpose_bf16_kernel<<<dim3((EMB_ * K1_) / 256, 1), 256, 0, stream>>>(
        Wt, WtT, K1_, EMB_, EMB_, 0, 0);
    transpose_bf16_kernel<<<dim3(((size_t)NCP_ * KB_) / 256, 1), 256, 0, stream>>>(
        Wb, WbT, KB_, NC_, NCP_, 0, 0);
    // 5. rel[b] = ht[b] @ seq[b]   (batched over grid.z)
    gemm_bf16_wmma<<<dim3(R_ / TM_, H_ / TN_, B_), 256, 0, stream>>>(
        ht, seqT, rel, nullptr, R_, H_, L_, H_, /*flags*/0,
        (long long)R_ * L_, (long long)H_ * L_, (long long)R_ * H_);
    // 6. A matrices = [gathered entity embeds | rel]
    build_A_kernel<<<((size_t)N_ * K1_) / 256, 256, 0, stream>>>(ent_embeds, rel, hts, Ah, At);
    // 7. extractors: tanh(A @ W + b) -> bf16
    gemm_bf16_wmma<<<dim3(N_ / TM_, EMB_ / TN_, 1), 256, 0, stream>>>(
        Ah, WhT, hsO, bh, N_, EMB_, K1_, EMB_, /*flags tanh*/2, 0, 0, 0);
    gemm_bf16_wmma<<<dim3(N_ / TM_, EMB_ / TN_, 1), 256, 0, stream>>>(
        At, WtT, tsO, bt, N_, EMB_, K1_, EMB_, /*flags tanh*/2, 0, 0, 0);
    // 8. block-bilinear classifier (A formed on the fly), f32 logits
    gemm_bilinear_wmma<<<dim3(N_ / TM_, NCP_ / TN_), 256, 0, stream>>>(hsO, tsO, WbT, bb, out);
}
